// CalculateAttention_9955734192580
// MI455X (gfx1250) — matrix operand order repaired and checked by reading, not compiled, not run on hardware
//
#include <hip/hip_runtime.h>

// Flash attention for B=2, H=8, S=4096, D=64 (fp32 in/out) on gfx1250 (MI455X).
// bf16 WMMA (v_wmma_f32_16x16x32_bf16) fp32-accum + base-2 online softmax.
// K/V (and transposed mask) pre-processed into workspace; tiles double-buffered
// in LDS via global_load_async_to_lds_b128 (ASYNCcnt, pipelined wait 0x4).

#define S_LEN 4096
#define D_HEAD 64
#define H_NUM 8
#define B_NUM 2

typedef __attribute__((ext_vector_type(16))) __bf16 v16bf;
typedef __attribute__((ext_vector_type(8)))  float  v8f;

union Frag {
    v16bf v;
    unsigned int   w[8];
    unsigned short u[16];
};

// base-2 exponential: v_exp_f32 directly (no log2e multiply)
#define EXP2F(x) __builtin_amdgcn_exp2f(x)

// packed f32x2 -> bf16x2 (RNE) in one VALU op
__device__ __forceinline__ unsigned cvt_pk_bf16(float a, float b) {
    unsigned d;
    asm("v_cvt_pk_bf16_f32 %0, %1, %2" : "=v"(d) : "v"(a), "v"(b));
    return d;   // [15:0]=bf16(a), [31:16]=bf16(b)
}

// CDNA5 async global->LDS copy, 16B per lane, tracked by ASYNCcnt.
__device__ __forceinline__ void async_ld_b128(unsigned lds_off, const void* gptr) {
    unsigned long long ga = (unsigned long long)gptr;
    asm volatile("global_load_async_to_lds_b128 %0, %1, off"
                 :: "v"(lds_off), "v"(ga) : "memory");
}
__device__ __forceinline__ void wait_async4() {
    asm volatile("s_wait_asynccnt 0x4" ::: "memory");
}
__device__ __forceinline__ void wait_async0() {
    asm volatile("s_wait_asynccnt 0x0" ::: "memory");
}
__device__ __forceinline__ unsigned lds_addr32(const void* p) {
    return (unsigned)(unsigned long long)p;   // low 32 bits = LDS offset
}

// LDS strides (ushort elements unless noted); rows are 16B multiples for async.
#define LDK 72    // K tile  [32 keys][64 d]    row = 144 B
#define LDV 40    // Vt tile [64 d][32 keys]    row = 80 B
#define LDP 36    // P scratch [16 rows][32 k]  per wave
#define LDM 32    // mask row tile [256 q][32 k] bytes (row-major path)
#define LDMT 272  // maskT tile [32 keys][256 q] bytes, padded (16B mult, 68 banks)

// base-2 softmax domain: scores carry 0.125*log2(e); mask fill = -1e9*0.125*log2(e)
#define QSCALE  0.180336880111102f       /* 0.125 * log2(e) */
#define MASKVAL -1.80336880111102e8f     /* -1e9 * 0.125 * log2(e) */

// ---------------- one-time preprocessing ----------------
__global__ __launch_bounds__(256) void prep_k_kernel(const float* __restrict__ K,
                                                     unsigned short* __restrict__ Kb) {
    const size_t i = ((size_t)blockIdx.x * 256 + threadIdx.x) * 8;
    float4 a = *(const float4*)(K + i);
    float4 b = *(const float4*)(K + i + 4);
    uint4 o;
    o.x = cvt_pk_bf16(a.x, a.y); o.y = cvt_pk_bf16(a.z, a.w);
    o.z = cvt_pk_bf16(b.x, b.y); o.w = cvt_pk_bf16(b.z, b.w);
    *(uint4*)(Kb + i) = o;
}

__global__ __launch_bounds__(256) void prep_vt_kernel(const float* __restrict__ V,
                                                      unsigned short* __restrict__ Vt) {
    const int t   = blockIdx.x * 256 + threadIdx.x;   // B*H*S*8 threads
    const int d0  = (t & 7) * 8;
    const int key = (t >> 3) & (S_LEN - 1);
    const int bh  = t >> 15;                           // /(8*4096)
    const float* src = V + ((size_t)bh * S_LEN + key) * D_HEAD + d0;
    unsigned short* dst = Vt + (size_t)bh * D_HEAD * S_LEN + key;
#pragma unroll
    for (int j = 0; j < 8; j += 2) {
        unsigned pk = cvt_pk_bf16(src[j], src[j + 1]);
        dst[(size_t)(d0 + j)     * S_LEN] = (unsigned short)pk;
        dst[(size_t)(d0 + j + 1) * S_LEN] = (unsigned short)(pk >> 16);
    }
}

// mask[q][k] (bytes) -> maskT[k][q] (bytes); shared across all (b,h)
__global__ __launch_bounds__(256) void prep_mt_kernel(const unsigned char* __restrict__ M,
                                                      unsigned char* __restrict__ MT) {
    const int c   = blockIdx.x * 256 + threadIdx.x;   // S*S/16 threads
    const int key = c & (S_LEN - 1);
    const int q0  = (c >> 12) * 16;
    unsigned b[16];
#pragma unroll
    for (int j = 0; j < 16; ++j) b[j] = M[(size_t)(q0 + j) * S_LEN + key];
    uint4 o;
    o.x = b[0]  | (b[1]  << 8) | (b[2]  << 16) | (b[3]  << 24);
    o.y = b[4]  | (b[5]  << 8) | (b[6]  << 16) | (b[7]  << 24);
    o.z = b[8]  | (b[9]  << 8) | (b[10] << 16) | (b[11] << 24);
    o.w = b[12] | (b[13] << 8) | (b[14] << 16) | (b[15] << 24);
    *(uint4*)(MT + (size_t)key * S_LEN + q0) = o;
}

// ---------------- main flash-attention kernel ----------------
__global__ __launch_bounds__(256)
void fa_fwd_kernel(const float* __restrict__ Q,
                   const float* __restrict__ Kf,
                   const float* __restrict__ Vf,
                   const unsigned short* __restrict__ Kb,
                   const unsigned short* __restrict__ Vtb,
                   const unsigned char* __restrict__ mask,
                   const unsigned char* __restrict__ maskT,
                   float* __restrict__ O,
                   int use_pre, int use_mt)
{
    __shared__ __align__(16) unsigned short lds_k[2][32 * LDK];
    __shared__ __align__(16) unsigned short lds_vt[2][D_HEAD * LDV];
    __shared__ __align__(16) unsigned char  lds_m[2][32 * LDMT];  // fits both layouts
    __shared__ unsigned short lds_p[8][16 * LDP];

    const int tid  = threadIdx.x;
    const int wave = tid >> 5;
    const int lane = tid & 31;
    const int half = lane >> 4;
    const int ln   = lane & 15;

    const int q0blk = blockIdx.x * 256;            // 8 waves x 32 rows
    const int q0    = q0blk + wave * 32;
    const size_t bh = ((size_t)blockIdx.z * H_NUM + blockIdx.y) * (size_t)S_LEN * D_HEAD;

    // ---- Q: two 16-row A-tiles x two 32-deep chunks, pre-scaled (base-2 domain) ----
    Frag aq[2][2];
#pragma unroll
    for (int mt = 0; mt < 2; ++mt) {
        const float* qsrc = Q + bh + (size_t)(q0 + mt * 16 + ln) * D_HEAD;
#pragma unroll
        for (int c = 0; c < 2; ++c)
#pragma unroll
            for (int v = 0; v < 8; ++v) {
                int kb = (v & 3) * 2 + half * 8 + ((v >> 2) << 4) + c * 32;
                aq[mt][c].w[v] = cvt_pk_bf16(qsrc[kb] * QSCALE, qsrc[kb + 1] * QSCALE);
            }
    }

    float m_run[2][8], l_run[2][8];
    v8f o[2][4];
#pragma unroll
    for (int mt = 0; mt < 2; ++mt) {
#pragma unroll
        for (int r = 0; r < 8; ++r) { m_run[mt][r] = -3.0e38f; l_run[mt][r] = 0.0f; }
#pragma unroll
        for (int nt = 0; nt < 4; ++nt) o[mt][nt] = (v8f){0,0,0,0,0,0,0,0};
    }

    // per-thread staging coordinates (within one buffer)
    const int krow = tid >> 3, kseg = tid & 7;       // K: 32 rows x 8 chunks
    const int vrow = tid >> 2, vseg = tid & 3;       // Vt: 64 rows x 4 chunks
    const int mkey = tid >> 3, mseg = (tid & 7) * 2; // maskT: 32 rows x 16 chunks (2/thread)
    const unsigned k_off  = (unsigned)(krow * LDK * 2 + kseg * 16);
    const unsigned v_off  = (unsigned)(vrow * LDV * 2 + vseg * 16);
    const unsigned m_off  = (unsigned)(tid * LDM);
    const unsigned mt_off = (unsigned)(mkey * LDMT + mseg * 16);

#define ISSUE_TILE(KTN, BUF)                                                              \
    do {                                                                                  \
        async_ld_b128(lds_addr32(&lds_k[BUF][0]) + k_off,                                 \
                      Kb + bh + (size_t)((KTN) + krow) * D_HEAD + kseg * 8);              \
        async_ld_b128(lds_addr32(&lds_vt[BUF][0]) + v_off,                                \
                      Vtb + bh + (size_t)vrow * S_LEN + (KTN) + vseg * 8);                \
        if (use_mt) {                                                                     \
            const unsigned char* mt_ = maskT + (size_t)((KTN) + mkey) * S_LEN             \
                                             + q0blk + mseg * 16;                         \
            async_ld_b128(lds_addr32(&lds_m[BUF][0]) + mt_off,      mt_);                 \
            async_ld_b128(lds_addr32(&lds_m[BUF][0]) + mt_off + 16, mt_ + 16);            \
        } else {                                                                          \
            const unsigned char* mr_ = mask + (size_t)(q0blk + tid) * S_LEN + (KTN);      \
            async_ld_b128(lds_addr32(&lds_m[BUF][0]) + m_off,      mr_);                  \
            async_ld_b128(lds_addr32(&lds_m[BUF][0]) + m_off + 16, mr_ + 16);             \
        }                                                                                 \
    } while (0)

    if (use_pre) { ISSUE_TILE(0, 0); }

    for (int kt = 0; kt < S_LEN; kt += 32) {
        const int ib = (kt >> 5) & 1;
        if (use_pre) {
            __syncthreads();                       // all waves done reading buf ib^1
            if (kt + 32 < S_LEN) { ISSUE_TILE(kt + 32, ib ^ 1); wait_async4(); }
            else                 { wait_async0(); }
            __syncthreads();                       // tile kt visible to all waves
        } else {
            __syncthreads();
            const int d0 = kseg * 8;
            const float* ksrc = Kf + bh + (size_t)(kt + krow) * D_HEAD + d0;
            const float* vsrc = Vf + bh + (size_t)(kt + krow) * D_HEAD + d0;
#pragma unroll
            for (int j = 0; j < 8; j += 2) {
                *(unsigned*)&lds_k[ib][krow * LDK + d0 + j] = cvt_pk_bf16(ksrc[j], ksrc[j + 1]);
                unsigned pv = cvt_pk_bf16(vsrc[j], vsrc[j + 1]);
                lds_vt[ib][(d0 + j)     * LDV + krow] = (unsigned short)pv;
                lds_vt[ib][(d0 + j + 1) * LDV + krow] = (unsigned short)(pv >> 16);
            }
            const unsigned char* mrow = mask + (size_t)(q0blk + tid) * S_LEN + kt;
            *(uint4*)(lds_m[ib] + tid * LDM)      = *(const uint4*)mrow;
            *(uint4*)(lds_m[ib] + tid * LDM + 16) = *(const uint4*)(mrow + 16);
            __syncthreads();
        }

        // ---- scores for both M-tiles: S = (Q*scale)(32x64) * K^T(64x32) ----
        v8f acc[2][2];
        acc[0][0] = (v8f){0,0,0,0,0,0,0,0}; acc[0][1] = acc[0][0];
        acc[1][0] = acc[0][0];              acc[1][1] = acc[0][0];
#pragma unroll
        for (int nt = 0; nt < 2; ++nt)
#pragma unroll
            for (int c = 0; c < 2; ++c) {
                Frag bk;
#pragma unroll
                for (int v = 0; v < 8; ++v)
                    bk.w[v] = *(const unsigned int*)
                        &lds_k[ib][(nt * 16 + ln) * LDK + c * 32 + (v & 3) * 2 + half * 8 + ((v >> 2) << 4)];
#pragma unroll
                for (int mt = 0; mt < 2; ++mt)
                    acc[mt][nt] = __builtin_amdgcn_wmma_f32_16x16x32_bf16(
                        false, aq[mt][c].v, false, bk.v, (short)0, acc[mt][nt], false, false);
            }

#pragma unroll
        for (int mt = 0; mt < 2; ++mt) {
            // ---- mask fill (base-2 domain constant) ----
            float s0[8], s1[8];
            if (use_mt) {
                // transposed tile: one b64 per score column -> no load chains
                const int qb = wave * 32 + mt * 16 + 8 * half;
                unsigned long long g0 = *(const unsigned long long*)&lds_m[ib][ln * LDMT + qb];
                unsigned long long g1 = *(const unsigned long long*)&lds_m[ib][(16 + ln) * LDMT + qb];
                unsigned lo0 = (unsigned)g0, hi0 = (unsigned)(g0 >> 32);
                unsigned lo1 = (unsigned)g1, hi1 = (unsigned)(g1 >> 32);
#pragma unroll
                for (int r = 0; r < 8; ++r) {
                    unsigned b0 = ((r < 4 ? lo0 : hi0) >> (8 * (r & 3))) & 0xffu;
                    unsigned b1 = ((r < 4 ? lo1 : hi1) >> (8 * (r & 3))) & 0xffu;
                    s0[r] = b0 ? MASKVAL : acc[mt][0][r];
                    s1[r] = b1 ? MASKVAL : acc[mt][1][r];
                }
            } else {
#pragma unroll
                for (int r = 0; r < 8; ++r) {
                    int qrow = wave * 32 + mt * 16 + r + half * 8;
                    unsigned char mk0 = lds_m[ib][qrow * LDM + ln];
                    unsigned char mk1 = lds_m[ib][qrow * LDM + 16 + ln];
                    s0[r] = mk0 ? MASKVAL : acc[mt][0][r];
                    s1[r] = mk1 ? MASKVAL : acc[mt][1][r];
                }
            }

            // ---- online softmax in exp2 domain (reduce across 16-lane halves) ----
            float mt8[8];
#pragma unroll
            for (int r = 0; r < 8; ++r) mt8[r] = fmaxf(s0[r], s1[r]);
#pragma unroll
            for (int off = 1; off < 16; off <<= 1)
#pragma unroll
                for (int r = 0; r < 8; ++r) mt8[r] = fmaxf(mt8[r], __shfl_xor(mt8[r], off, 32));

            float corr[8], rs[8];
#pragma unroll
            for (int r = 0; r < 8; ++r) {
                float mnew = fmaxf(m_run[mt][r], mt8[r]);
                corr[r] = EXP2F(m_run[mt][r] - mnew);
                m_run[mt][r] = mnew;
                float p0 = EXP2F(s0[r] - mnew);
                float p1 = EXP2F(s1[r] - mnew);
                s0[r] = p0; s1[r] = p1;
                rs[r] = p0 + p1;
            }
#pragma unroll
            for (int off = 1; off < 16; off <<= 1)
#pragma unroll
                for (int r = 0; r < 8; ++r) rs[r] += __shfl_xor(rs[r], off, 32);
#pragma unroll
            for (int r = 0; r < 8; ++r) {
                l_run[mt][r] = l_run[mt][r] * corr[r] + rs[r];
                o[mt][0][r] *= corr[r]; o[mt][1][r] *= corr[r];
                o[mt][2][r] *= corr[r]; o[mt][3][r] *= corr[r];
            }

            // ---- transpose P (C layout -> A layout) via per-wave LDS scratch ----
#pragma unroll
            for (int r = 0; r < 8; ++r) {
                int prow = r + 8 * half;
                unsigned pk = cvt_pk_bf16(s0[r], s1[r]);
                lds_p[wave][prow * LDP + ln]      = (unsigned short)pk;
                lds_p[wave][prow * LDP + 16 + ln] = (unsigned short)(pk >> 16);
            }
            Frag ap;
#pragma unroll
            for (int v = 0; v < 8; ++v) {
                int kb = (v & 3) * 2 + half * 8 + ((v >> 2) << 4);
                ap.w[v] = *(const unsigned int*)&lds_p[wave][ln * LDP + kb];
            }

            // ---- O += P(16x32) * V(32x64)  (bv re-read per M-tile: lower VGPR pressure) ----
#pragma unroll
            for (int nt = 0; nt < 4; ++nt) {
                Frag bv;
#pragma unroll
                for (int v = 0; v < 8; ++v)
                    bv.w[v] = *(const unsigned int*)
                        &lds_vt[ib][(nt * 16 + ln) * LDV + (v & 3) * 2 + half * 8 + ((v >> 2) << 4)];
                o[mt][nt] = __builtin_amdgcn_wmma_f32_16x16x32_bf16(
                    false, ap.v, false, bv.v, (short)0, o[mt][nt], false, false);
            }
        }
    }

    // ---- normalize + store ----
#pragma unroll
    for (int mt = 0; mt < 2; ++mt)
#pragma unroll
        for (int r = 0; r < 8; ++r) {
            float inv = 1.0f / l_run[mt][r];
            size_t row = bh + (size_t)(q0 + mt * 16 + r + 8 * half) * D_HEAD;
            O[row +  0 + ln] = o[mt][0][r] * inv;
            O[row + 16 + ln] = o[mt][1][r] * inv;
            O[row + 32 + ln] = o[mt][2][r] * inv;
            O[row + 48 + ln] = o[mt][3][r] * inv;
        }
}

extern "C" void kernel_launch(void* const* d_in, const int* in_sizes, int n_in,
                              void* d_out, int out_size, void* d_ws, size_t ws_size,
                              hipStream_t stream) {
    const float* Q = (const float*)d_in[0];
    const float* K = (const float*)d_in[1];
    const float* V = (const float*)d_in[2];
    const unsigned char* mask = (const unsigned char*)d_in[3];  // jnp.bool_ -> 1 byte
    float* O = (float*)d_out;

    const size_t elems = (size_t)B_NUM * H_NUM * S_LEN * D_HEAD;   // 4,194,304
    const size_t kv_sz = elems * 2 * 2;                            // K+Vt bf16 = 16 MB
    const size_t mt_sz = (size_t)S_LEN * S_LEN;                    // maskT     = 16 MB
    const int use_pre = (ws_size >= kv_sz) ? 1 : 0;
    const int use_mt  = (ws_size >= kv_sz + mt_sz) ? 1 : 0;

    unsigned short* Kb  = (unsigned short*)d_ws;
    unsigned short* Vtb = (unsigned short*)((char*)d_ws + elems * 2);
    unsigned char*  MT  = (unsigned char*)((char*)d_ws + kv_sz);

    if (use_pre) {
        prep_k_kernel<<<dim3((unsigned)(elems / (256 * 8))), 256, 0, stream>>>(K, Kb);
        prep_vt_kernel<<<dim3((unsigned)(elems / (256 * 8))), 256, 0, stream>>>(V, Vtb);
    }
    if (use_mt) {
        prep_mt_kernel<<<dim3((unsigned)((size_t)S_LEN * S_LEN / (256 * 16))), 256, 0, stream>>>(mask, MT);
    }
    dim3 grid(S_LEN / 256, H_NUM, B_NUM);
    fa_fwd_kernel<<<grid, 256, 0, stream>>>(Q, K, V, Kb, Vtb, mask, MT, O, use_pre, use_mt);
}
